// MultiHeadSelfAttention_26190710571253
// MI455X (gfx1250) — compile-verified, matrix operand order checked by reading, not run on hardware
//
#include <hip/hip_runtime.h>

// ---------------------------------------------------------------------------
// MultiHeadSelfAttention forward for MI455X (gfx1250, wave32, WMMA).
// B=4, S=2048, D=1024, H=16, HD=64.  ~137 GFLOP vs ~100MB traffic -> compute
// bound -> all matmuls via v_wmma_f32_16x16x32_bf16 (fp32 accum).
// Workspace (64 MB): Q[B,H,S,HD] | K[B,H,S,HD] | V[B,H,HD,S] | O[B*S,D], bf16.
// Kernel 3 stages its bf16 A-tile with the Tensor Data Mover (TDM) and
// s_wait_tensorcnt -- the CDNA5 async-tensor path.
// ---------------------------------------------------------------------------

typedef __attribute__((ext_vector_type(16))) __bf16 v16bf;
typedef __attribute__((ext_vector_type(8)))  float  v8f;
typedef __attribute__((ext_vector_type(4)))  unsigned int v4u;
typedef __attribute__((ext_vector_type(8)))  int    v8i;
typedef __attribute__((ext_vector_type(4)))  int    v4i;

union Frag16 { v16bf v; uint4 q[2]; };

// Native fp32 -> bf16 (RTNE) via hardware convert.
__device__ __forceinline__ unsigned short f2bf(float f) {
  union { __bf16 h; unsigned short u; } c;
  c.h = (__bf16)f;
  return c.u;
}

// Load a 16x32 bf16 WMMA operand fragment (A layout; B mirrored lane=column).
//   lanes 0-15 : row = lane,    K in {0..7, 16..23}
//   lanes 16-31: row = lane-16, K in {8..15, 24..31}
// Two 16-byte loads per lane.  `p` -> element (row0, k0), row stride `ld`.
__device__ __forceinline__ v16bf ld_frag(const unsigned short* p, int ld, int lane) {
  const int half = lane >> 4;
  const int r    = lane & 15;
  const unsigned short* rp = p + r * ld + half * 8;
  Frag16 f;
  f.q[0] = *(const uint4*)(rp);
  f.q[1] = *(const uint4*)(rp + 16);
  return f.v;
}

__device__ __forceinline__ v8f wmma_bf16(v16bf a, v16bf b, v8f c) {
  return __builtin_amdgcn_wmma_f32_16x16x32_bf16(false, a, false, b,
                                                 (short)0, c, false, false);
}

#define LDT 40   // padded LDS row stride (elements): 80B rows, 16B aligned

// ---------------------------------------------------------------------------
// Kernel 1: QKV projection.  X[8192,1024] @ W[1024,3072] + b -> Q/K/V (bf16).
// Block tile 64x128, 4 waves (2x2), wave tile 32x64 (2x4 frags, 8 WMMA), BK=32.
// ---------------------------------------------------------------------------
__global__ __launch_bounds__(128) void qkv_gemm_kernel(
    const float* __restrict__ X, const float* __restrict__ W,
    const float* __restrict__ bias,
    unsigned short* __restrict__ Qo, unsigned short* __restrict__ Ko,
    unsigned short* __restrict__ Vo)
{
  const int Kd = 1024, Nn = 3072;
  __shared__ unsigned short As[64  * LDT];   // [m][k] bf16
  __shared__ unsigned short Bs[128 * LDT];   // [n][k] bf16 (transposed W tile)

  const int tid  = threadIdx.x;
  const int lane = tid & 31;
  const int wave = tid >> 5;
  const int bm   = blockIdx.y * 64;
  const int bn   = blockIdx.x * 128;
  const int wm   = (wave >> 1) * 32;
  const int wn   = (wave & 1)  * 64;

  v8f acc[2][4] = {{{}, {}, {}, {}}, {{}, {}, {}, {}}};

  for (int kk = 0; kk < Kd; kk += 32) {
    // Stage A: 64x32 fp32 -> bf16, row-major.
    for (int idx = tid; idx < 512; idx += 128) {
      const int row = idx >> 3, c4 = (idx & 7) * 4;
      const float4 f = *(const float4*)(X + (size_t)(bm + row) * Kd + kk + c4);
      uint2 pk;
      pk.x = (unsigned)f2bf(f.x) | ((unsigned)f2bf(f.y) << 16);
      pk.y = (unsigned)f2bf(f.z) | ((unsigned)f2bf(f.w) << 16);
      *(uint2*)(As + row * LDT + c4) = pk;
    }
    // Stage B transposed: Bs[n][k] = W[kk+k][bn+n].   32k x 128n elements.
    for (int idx = tid; idx < 1024; idx += 128) {
      const int k = idx >> 5, n4 = (idx & 31) * 4;
      const float4 f = *(const float4*)(W + (size_t)(kk + k) * Nn + bn + n4);
      Bs[(n4 + 0) * LDT + k] = f2bf(f.x);
      Bs[(n4 + 1) * LDT + k] = f2bf(f.y);
      Bs[(n4 + 2) * LDT + k] = f2bf(f.z);
      Bs[(n4 + 3) * LDT + k] = f2bf(f.w);
    }
    __syncthreads();

    const v16bf a0 = ld_frag(As + (wm +  0) * LDT, LDT, lane);
    const v16bf a1 = ld_frag(As + (wm + 16) * LDT, LDT, lane);
    #pragma unroll
    for (int j = 0; j < 4; ++j) {
      const v16bf bf = ld_frag(Bs + (wn + j * 16) * LDT, LDT, lane);
      acc[0][j] = wmma_bf16(a0, bf, acc[0][j]);
      acc[1][j] = wmma_bf16(a1, bf, acc[1][j]);
    }
    __syncthreads();
  }

  // Epilogue: bias add, split into Q [B,H,S,HD], K [B,H,S,HD], V [B,H,HD,S].
  const int half = lane >> 4, nl = lane & 15;
  #pragma unroll
  for (int i = 0; i < 2; ++i) {
    #pragma unroll
    for (int j = 0; j < 4; ++j) {
      const int n   = bn + wn + j * 16 + nl;
      const float bv = bias[n];
      const int reg = n >> 10;          // 0:Q 1:K 2:V
      const int rem = n & 1023;
      const int h = rem >> 6, d = rem & 63;
      #pragma unroll
      for (int r = 0; r < 8; ++r) {
        const int m = bm + wm + i * 16 + r + half * 8;
        const int b = m >> 11, s = m & 2047;
        const unsigned short val = f2bf(acc[i][j][r] + bv);
        if (reg == 0)
          Qo[((size_t)(b * 16 + h) * 2048 + s) * 64 + d] = val;
        else if (reg == 1)
          Ko[((size_t)(b * 16 + h) * 2048 + s) * 64 + d] = val;
        else
          Vo[((size_t)(b * 16 + h) * 64 + d) * 2048 + s] = val;  // transposed
      }
    }
  }
}

// ---------------------------------------------------------------------------
// Kernel 2: causal flash attention.  One wave = 16 query rows; key tiles of
// 32; online softmax; P routed C-layout -> A-layout via per-wave LDS slab.
// ---------------------------------------------------------------------------
#define LDP 40

__global__ __launch_bounds__(128) void flash_attn_kernel(
    const unsigned short* __restrict__ Q,   // [B,H,S,HD] bf16
    const unsigned short* __restrict__ Kt,  // [B,H,S,HD] bf16
    const unsigned short* __restrict__ Vt,  // [B,H,HD,S] bf16 (pre-transposed)
    unsigned short* __restrict__ O)         // [B*S, D]   bf16
{
  const int S = 2048, HD = 64;
  __shared__ unsigned short Pl[4][16 * LDP];

  const int lane = threadIdx.x & 31;
  const int wave = threadIdx.x >> 5;
  const int bh   = blockIdx.x >> 5;
  const int qb   = blockIdx.x & 31;
  const int q0   = qb * 64 + wave * 16;
  const int half = lane >> 4, nl = lane & 15;

  const unsigned short* Qb = Q  + (size_t)bh * S * HD;
  const unsigned short* Kb = Kt + (size_t)bh * S * HD;
  const unsigned short* Vb = Vt + (size_t)bh * HD * S;
  unsigned short* Pw = &Pl[wave][0];

  const v16bf aq0 = ld_frag(Qb + (size_t)q0 * HD +  0, HD, lane);
  const v16bf aq1 = ld_frag(Qb + (size_t)q0 * HD + 32, HD, lane);

  v8f o0 = {}, o1 = {}, o2 = {}, o3 = {};
  float mi[8], li[8];
  #pragma unroll
  for (int r = 0; r < 8; ++r) { mi[r] = -3.0e38f; li[r] = 0.0f; }

  const int nkt = (q0 + 47) >> 5;   // ceil((q0+16)/32) causal key tiles
  for (int kt = 0; kt < nkt; ++kt) {
    const int k0 = kt * 32;

    v8f sc0 = {}, sc1 = {};
    {
      const v16bf b00 = ld_frag(Kb + (size_t)(k0     ) * HD +  0, HD, lane);
      const v16bf b01 = ld_frag(Kb + (size_t)(k0     ) * HD + 32, HD, lane);
      sc0 = wmma_bf16(aq0, b00, sc0);
      sc0 = wmma_bf16(aq1, b01, sc0);
      const v16bf b10 = ld_frag(Kb + (size_t)(k0 + 16) * HD +  0, HD, lane);
      const v16bf b11 = ld_frag(Kb + (size_t)(k0 + 16) * HD + 32, HD, lane);
      sc1 = wmma_bf16(aq0, b10, sc1);
      sc1 = wmma_bf16(aq1, b11, sc1);
    }

    #pragma unroll
    for (int r = 0; r < 8; ++r) {
      const int row = q0 + r + half * 8;
      float s0 = sc0[r] * 0.125f;             // 1/sqrt(64)
      float s1 = sc1[r] * 0.125f;
      if (k0 + nl      > row) s0 = -3.0e38f;
      if (k0 + 16 + nl > row) s1 = -3.0e38f;

      float vmax = fmaxf(s0, s1);
      #pragma unroll
      for (int m = 1; m < 16; m <<= 1) vmax = fmaxf(vmax, __shfl_xor(vmax, m, 32));
      const float nm    = fmaxf(mi[r], vmax);
      const float alpha = __expf(mi[r] - nm);
      mi[r] = nm;

      const float p0 = __expf(s0 - nm);
      const float p1 = __expf(s1 - nm);
      float rs = p0 + p1;
      #pragma unroll
      for (int m = 1; m < 16; m <<= 1) rs += __shfl_xor(rs, m, 32);
      li[r] = li[r] * alpha + rs;

      o0[r] *= alpha; o1[r] *= alpha; o2[r] *= alpha; o3[r] *= alpha;

      Pw[(r + half * 8) * LDP + nl     ] = f2bf(p0);
      Pw[(r + half * 8) * LDP + 16 + nl] = f2bf(p1);
    }

    const v16bf ap  = ld_frag(Pw, LDP, lane);
    const v16bf bv0 = ld_frag(Vb + (size_t) 0 * S + k0, S, lane);
    const v16bf bv1 = ld_frag(Vb + (size_t)16 * S + k0, S, lane);
    const v16bf bv2 = ld_frag(Vb + (size_t)32 * S + k0, S, lane);
    const v16bf bv3 = ld_frag(Vb + (size_t)48 * S + k0, S, lane);
    o0 = wmma_bf16(ap, bv0, o0);
    o1 = wmma_bf16(ap, bv1, o1);
    o2 = wmma_bf16(ap, bv2, o2);
    o3 = wmma_bf16(ap, bv3, o3);
  }

  const int b = bh >> 4, h = bh & 15;
  #pragma unroll
  for (int r = 0; r < 8; ++r) {
    const int   t   = q0 + r + half * 8;
    const float inv = 1.0f / li[r];
    const size_t base = ((size_t)(b * 2048 + t) * 1024) + h * 64 + nl;
    O[base +  0] = f2bf(o0[r] * inv);
    O[base + 16] = f2bf(o1[r] * inv);
    O[base + 32] = f2bf(o2[r] * inv);
    O[base + 48] = f2bf(o3[r] * inv);
  }
}

// ---------------------------------------------------------------------------
// Kernel 3: output projection.  O_bf16[8192,1024] @ W_out[1024,1024] + b
// -> fp32 d_out.  Block tile 64x128, wave tile 32x64.  The bf16 A-tile is
// staged with the Tensor Data Mover: a 2D D# descriptor (ISA ch.8) with
// data_size=2B, tile 32x64, and LDS padding (pad_interval=16 DWORDs,
// pad_amount=4 DWORDs) reproducing the LDT=40 padded layout directly.
// ---------------------------------------------------------------------------
__global__ __launch_bounds__(128) void out_gemm_kernel(
    const unsigned short* __restrict__ A, const float* __restrict__ W,
    const float* __restrict__ bias, float* __restrict__ out)
{
  const int Kd = 1024, Nn = 1024;
  __shared__ unsigned short As[64  * LDT];
  __shared__ unsigned short Bs[128 * LDT];

  const int tid  = threadIdx.x;
  const int lane = tid & 31;
  const int wave = tid >> 5;
  const int bm   = blockIdx.y * 64;
  const int bn   = blockIdx.x * 128;
  const int wm   = (wave >> 1) * 32;
  const int wn   = (wave & 1)  * 64;

  v8f acc[2][4] = {{{}, {}, {}, {}}, {{}, {}, {}, {}}};

  // Invariant pieces of the TDM descriptor (D#).
  const unsigned ldsAddr = (unsigned)(unsigned long long)(As);  // LDS byte addr
  v8i g1;
  g1[0] = (int)((1u << 16)        // data_size = 1 -> 2-byte elements
              | (1u << 20)        // pad_enable
              | (3u << 22)        // pad_interval: 16 DWORDs (one 32-elem row)
              | (3u << 25));      // pad_amount:   4 DWORDs (8 elems -> LDT=40)
  g1[1] = (int)(((unsigned)Kd & 0xFFFFu) << 16);          // tensor_dim0 lo16
  g1[2] = (int)(((unsigned)Kd >> 16) | (8192u << 16));    // dim0 hi | dim1 lo
  g1[3] = (int)((8192u >> 16) | (32u << 16));             // dim1 hi | tile_dim0
  g1[4] = (int)(64u);                                     // tile_dim1 (tile_dim2=0)
  g1[5] = (int)((unsigned)Kd);                            // tensor_dim0_stride lo
  g1[6] = 0;                                              // stride0 hi | stride1 lo
  g1[7] = 0;
  const v4i gz4 = {0, 0, 0, 0};
  const v8i gz8 = {0, 0, 0, 0, 0, 0, 0, 0};

  for (int kk = 0; kk < Kd; kk += 32) {
    // A-tile via TDM (wave 0 issues; EXEC is ignored by tensor ops).
    if (tid < 32) {
      const unsigned long long ga =
          (unsigned long long)(const void*)(A + (size_t)bm * Kd + kk);
      v4u g0;
      g0[0] = 1u;                                   // count=1, user descriptor
      g0[1] = ldsAddr;                              // lds_addr
      g0[2] = (unsigned)(ga & 0xFFFFFFFFu);         // global_addr[31:0]
      g0[3] = (unsigned)((ga >> 32) & 0x01FFFFFFu)  // global_addr[56:32]
            | (2u << 30);                           // type = 2 (image)
      __builtin_amdgcn_tensor_load_to_lds(g0, g1, gz4, gz4, gz8, 0);
    }
    // Stage B transposed: Bs[n][k] = W[kk+k][bn+n].  (overlaps the TDM)
    for (int idx = tid; idx < 1024; idx += 128) {
      const int k = idx >> 5, n4 = (idx & 31) * 4;
      const float4 f = *(const float4*)(W + (size_t)(kk + k) * Nn + bn + n4);
      Bs[(n4 + 0) * LDT + k] = f2bf(f.x);
      Bs[(n4 + 1) * LDT + k] = f2bf(f.y);
      Bs[(n4 + 2) * LDT + k] = f2bf(f.z);
      Bs[(n4 + 3) * LDT + k] = f2bf(f.w);
    }
    if (tid < 32) __builtin_amdgcn_s_wait_tensorcnt(0);
    __syncthreads();

    const v16bf a0 = ld_frag(As + (wm +  0) * LDT, LDT, lane);
    const v16bf a1 = ld_frag(As + (wm + 16) * LDT, LDT, lane);
    #pragma unroll
    for (int j = 0; j < 4; ++j) {
      const v16bf bf = ld_frag(Bs + (wn + j * 16) * LDT, LDT, lane);
      acc[0][j] = wmma_bf16(a0, bf, acc[0][j]);
      acc[1][j] = wmma_bf16(a1, bf, acc[1][j]);
    }
    __syncthreads();
  }

  const int half = lane >> 4, nl = lane & 15;
  #pragma unroll
  for (int i = 0; i < 2; ++i) {
    #pragma unroll
    for (int j = 0; j < 4; ++j) {
      const int n = bn + wn + j * 16 + nl;
      const float bv = bias[n];
      #pragma unroll
      for (int r = 0; r < 8; ++r) {
        const int m = bm + wm + i * 16 + r + half * 8;
        out[(size_t)m * Nn + n] = acc[i][j][r] + bv;
      }
    }
  }
}

// ---------------------------------------------------------------------------
extern "C" void kernel_launch(void* const* d_in, const int* in_sizes, int n_in,
                              void* d_out, int out_size, void* d_ws, size_t ws_size,
                              hipStream_t stream) {
  // setup_inputs order: x, causal_mask, padding_mask, W_qkv, b_qkv, W_out, b_out
  const float* x    = (const float*)d_in[0];
  const float* Wqkv = (const float*)d_in[3];
  const float* bqkv = (const float*)d_in[4];
  const float* Wout = (const float*)d_in[5];
  const float* bout = (const float*)d_in[6];
  float* out = (float*)d_out;

  const size_t SEG = (size_t)8 * 1024 * 1024;   // 8Mi bf16 elements = 16 MB
  unsigned short* Qw = (unsigned short*)d_ws;
  unsigned short* Kw = Qw + SEG;
  unsigned short* Vw = Kw + SEG;
  unsigned short* Ow = Vw + SEG;

  qkv_gemm_kernel <<<dim3(3072 / 128, 8192 / 64), 128, 0, stream>>>(x, Wqkv, bqkv, Qw, Kw, Vw);
  flash_attn_kernel<<<dim3(64 * 32),              128, 0, stream>>>(Qw, Kw, Vw, Ow);
  out_gemm_kernel <<<dim3(1024 / 128, 8192 / 64), 128, 0, stream>>>(Ow, Wout, bout, out);
}
